// GatedFFN_5342939316974
// MI455X (gfx1250) — matrix operand order repaired and checked by reading, not compiled, not run on hardware
//
#include <hip/hip_runtime.h>
#include <cstdint>

// ---------------- types ----------------
typedef __bf16 bf16;
typedef bf16 bf16x8  __attribute__((ext_vector_type(8)));
typedef bf16 bf16x16 __attribute__((ext_vector_type(16)));
typedef float f32x8  __attribute__((ext_vector_type(8)));
typedef unsigned short u16;
typedef u16 u16x4 __attribute__((ext_vector_type(4)));
typedef u16 u16x8 __attribute__((ext_vector_type(8)));

// ---------------- problem constants ----------------
constexpr int BB = 2, TT = 1024, DMODEL = 1024, TILES = 8, DFF = 4096;
constexpr int NTOK = BB * TT;            // 2048 tokens
constexpr int NPAD = NTOK + TILES * 32;  // 2304 padded rows (32-row segs per expert)
constexpr int MT   = NPAD / 32;          // 72 M-blocks (32 rows each)

// ---------------- workspace layout (bytes) ----------------
constexpr size_t WS_COUNTS = 0;                               // 8 x i32
constexpr size_t WS_CURS   = 32;                              // 8 x i32
constexpr size_t WS_POFF   = 64;                              // 9 x i32
constexpr size_t WS_TIDX   = 128;                             // NTOK x i32
constexpr size_t WS_PERM   = WS_TIDX + (size_t)NTOK * 4;      // NPAD x i32 (ends 17536)
constexpr size_t WS_XP     = 18432;                           // NPAD x D bf16 (permuted x)
constexpr size_t WS_UPB    = WS_XP  + (size_t)NPAD * DMODEL * 2;        // up_w bf16 B-tiles
constexpr size_t WS_WDS    = WS_UPB + (size_t)TILES * DMODEL * DFF * 2; // sum(down_w) B-tiles
constexpr size_t WS_HID    = WS_WDS + (size_t)DFF * DMODEL * 2;         // NPAD x DFF bf16

__device__ __forceinline__ u16 f2bf(float f) {
    union { float f; uint32_t u; } v; v.f = f;
    uint32_t u = v.u;
    return (u16)((u + 0x7FFFu + ((u >> 16) & 1u)) >> 16);   // RNE
}

// ---------------- K0: init header ----------------
__global__ void k_init(char* ws) {
    int i = blockIdx.x * blockDim.x + threadIdx.x;
    int* perm = (int*)(ws + WS_PERM);
    if (i < NPAD) perm[i] = -1;
    if (i < TILES) {
        ((int*)(ws + WS_COUNTS))[i] = 0;
        ((int*)(ws + WS_CURS))[i]   = 0;
    }
}

// ---------------- K1: gating (one block per token) ----------------
__global__ __launch_bounds__(256) void k_gate(const float* __restrict__ x,
                                              const float* __restrict__ gw,
                                              const float* __restrict__ gb,
                                              char* ws, float* __restrict__ gate_out) {
    __shared__ float sred[256 * TILES];
    int n = blockIdx.x, tid = threadIdx.x;
    float a[TILES];
    #pragma unroll
    for (int t = 0; t < TILES; ++t) a[t] = 0.f;
    for (int d = tid; d < DMODEL; d += 256) {
        float xv = x[(size_t)n * DMODEL + d];
        #pragma unroll
        for (int t = 0; t < TILES; ++t) a[t] += xv * gw[d * TILES + t];
    }
    #pragma unroll
    for (int t = 0; t < TILES; ++t) sred[tid * TILES + t] = a[t];
    __syncthreads();
    for (int s = 128; s > 0; s >>= 1) {
        if (tid < s)
            #pragma unroll
            for (int t = 0; t < TILES; ++t) sred[tid * TILES + t] += sred[(tid + s) * TILES + t];
        __syncthreads();
    }
    if (tid == 0) {
        float best = sred[0] + gb[0]; int bi = 0;
        #pragma unroll
        for (int t = 1; t < TILES; ++t) {
            float l = sred[t] + gb[t];
            if (l > best) { best = l; bi = t; }   // first-max, matches jnp.argmax
        }
        ((int*)(ws + WS_TIDX))[n] = bi;
        atomicAdd((int*)(ws + WS_COUNTS) + bi, 1);
        #pragma unroll
        for (int t = 0; t < TILES; ++t) gate_out[(size_t)n * TILES + t] = (t == bi) ? 1.f : 0.f;
    }
}

// ---------------- K2: padded segment offsets (32-row granularity) ----------------
__global__ void k_offsets(char* ws) {
    const int* counts = (const int*)(ws + WS_COUNTS);
    int* poff = (int*)(ws + WS_POFF);
    int run = 0;
    for (int t = 0; t < TILES; ++t) { poff[t] = run; run += ((counts[t] + 31) >> 5) << 5; }
    poff[TILES] = run;
}

// ---------------- K3: scatter tokens into padded permutation ----------------
__global__ void k_scatter(char* ws) {
    int n = blockIdx.x * blockDim.x + threadIdx.x;
    if (n >= NTOK) return;
    int t = ((const int*)(ws + WS_TIDX))[n];
    int pos = atomicAdd((int*)(ws + WS_CURS) + t, 1);
    ((int*)(ws + WS_PERM))[((const int*)(ws + WS_POFF))[t] + pos] = n;
}

// ---------------- K4: build permuted bf16 activations (pad rows = 0) ----------------
__global__ __launch_bounds__(256) void k_xp(const float* __restrict__ x, char* ws) {
    int row = blockIdx.x, tid = threadIdx.x;
    int tok = ((const int*)(ws + WS_PERM))[row];
    u16x4* dst = (u16x4*)(ws + WS_XP + (size_t)row * DMODEL * 2);
    float4 v = make_float4(0.f, 0.f, 0.f, 0.f);
    if (tok >= 0) v = ((const float4*)(x + (size_t)tok * DMODEL))[tid];
    u16x4 o; o.x = f2bf(v.x); o.y = f2bf(v.y); o.z = f2bf(v.z); o.w = f2bf(v.w);
    dst[tid] = o;
}

// ---------------- K5: pack up_w fp32 -> bf16 WMMA B tiles ----------------
// tile(t,kb,nb) = 32x16 bf16, 1024B; lane L holds col n=L%16, K=(L/16)*16+i, i=0..15
__global__ __launch_bounds__(256) void k_pack_up(const float* __restrict__ w, char* ws) {
    size_t idx = (size_t)blockIdx.x * 256 + threadIdx.x;  // over TILES*(D/2)*DFF
    int n = (int)(idx % DFF);
    size_t rest = idx / DFF;
    int kp = (int)(rest % (DMODEL / 2));
    int t  = (int)(rest / (DMODEL / 2));
    int k = kp * 2;
    float w0 = w[((size_t)t * DMODEL + k) * DFF + n];
    float w1 = w[((size_t)t * DMODEL + k + 1) * DFF + n];
    uint32_t pk = (uint32_t)f2bf(w0) | ((uint32_t)f2bf(w1) << 16);
    int kb = k >> 5, kin = k & 31, nb = n >> 4, nin = n & 15;
    int lane = (kin >> 4) * 16 + nin;
    size_t tile = ((size_t)t * (DMODEL / 32) + kb) * (DFF / 16) + nb;
    *(uint32_t*)(ws + WS_UPB + tile * 1024 + lane * 32 + ((kin & 15) >> 1) * 4) = pk;
}

// ---------------- K6: pack sum_t(down_w) -> bf16 WMMA B tiles ----------------
__global__ __launch_bounds__(256) void k_pack_wds(const float* __restrict__ w, char* ws) {
    size_t idx = (size_t)blockIdx.x * 256 + threadIdx.x;  // over (DFF/2)*D
    int n  = (int)(idx % DMODEL);
    int kp = (int)(idx / DMODEL);
    int k = kp * 2;
    float s0 = 0.f, s1 = 0.f;
    #pragma unroll
    for (int t = 0; t < TILES; ++t) {
        s0 += w[((size_t)t * DFF + k)     * DMODEL + n];
        s1 += w[((size_t)t * DFF + k + 1) * DMODEL + n];
    }
    uint32_t pk = (uint32_t)f2bf(s0) | ((uint32_t)f2bf(s1) << 16);
    int kb = k >> 5, kin = k & 31, nb = n >> 4, nin = n & 15;
    int lane = (kin >> 4) * 16 + nin;
    size_t tile = (size_t)kb * (DMODEL / 16) + nb;
    *(uint32_t*)(ws + WS_WDS + tile * 1024 + lane * 32 + ((kin & 15) >> 1) * 4) = pk;
}

// ---------------- K7: up GEMM, 32x32 per wave (2x2 WMMA tiles), +bias, ReLU ----------
__global__ __launch_bounds__(128) void k_up_gemm(char* ws, const float* __restrict__ up_b) {
    __shared__ float sm[4][4][256];
    int lane = threadIdx.x & 31, w = threadIdx.x >> 5;
    int m = blockIdx.y;
    int nb0 = (blockIdx.x * 4 + w) * 2;           // 2 adjacent 16-col tiles per wave
    int row0 = m * 32;
    const int* poff = (const int*)(ws + WS_POFF);
    int e = 0;
    #pragma unroll
    for (int t = 1; t < TILES; ++t) if (row0 >= poff[t]) e = t;

    const bf16* A0 = (const bf16*)(ws + WS_XP) +
                     (size_t)(row0 + (lane & 15)) * DMODEL + (lane >> 4) * 16;
    const bf16* A1 = A0 + (size_t)16 * DMODEL;
    const bf16* Bp = (const bf16*)(ws + WS_UPB) +
                     (((size_t)e * (DMODEL / 32)) * (DFF / 16) + nb0) * 512 + lane * 16;
    constexpr size_t BSTRIDE = (size_t)(DFF / 16) * 512;   // elems per kb step

    f32x8 acc[2][2];
    #pragma unroll
    for (int i = 0; i < 2; ++i)
        #pragma unroll
        for (int j = 0; j < 2; ++j) acc[i][j] = (f32x8){0.f,0.f,0.f,0.f,0.f,0.f,0.f,0.f};

    #pragma unroll 2
    for (int kb = 0; kb < DMODEL / 32; ++kb) {
        bf16x8 a00 = *(const bf16x8*)(A0 + kb * 32);
        bf16x8 a01 = *(const bf16x8*)(A0 + kb * 32 + 8);
        bf16x8 a10 = *(const bf16x8*)(A1 + kb * 32);
        bf16x8 a11 = *(const bf16x8*)(A1 + kb * 32 + 8);
        bf16x16 af0 = __builtin_shufflevector(a00, a01, 0,1,2,3,4,5,6,7,8,9,10,11,12,13,14,15);
        bf16x16 af1 = __builtin_shufflevector(a10, a11, 0,1,2,3,4,5,6,7,8,9,10,11,12,13,14,15);
        bf16x16 bf0 = *(const bf16x16*)(Bp + kb * BSTRIDE);
        bf16x16 bf1 = *(const bf16x16*)(Bp + kb * BSTRIDE + 512);
        acc[0][0] = __builtin_amdgcn_wmma_f32_16x16x32_bf16(false, af0, false, bf0, (short)0, acc[0][0], false, false);
        acc[0][1] = __builtin_amdgcn_wmma_f32_16x16x32_bf16(false, af0, false, bf1, (short)0, acc[0][1], false, false);
        acc[1][0] = __builtin_amdgcn_wmma_f32_16x16x32_bf16(false, af1, false, bf0, (short)0, acc[1][0], false, false);
        acc[1][1] = __builtin_amdgcn_wmma_f32_16x16x32_bf16(false, af1, false, bf1, (short)0, acc[1][1], false, false);
    }

    float bias[2];
    #pragma unroll
    for (int j = 0; j < 2; ++j)
        bias[j] = up_b[(size_t)e * DFF + (nb0 + j) * 16 + (lane & 15)];

    #pragma unroll
    for (int i = 0; i < 2; ++i)
        #pragma unroll
        for (int j = 0; j < 2; ++j)
            #pragma unroll
            for (int r = 0; r < 8; ++r) {
                float v = acc[i][j][r] + bias[j];
                sm[w][i * 2 + j][(r + 8 * (lane >> 4)) * 16 + (lane & 15)] = v > 0.f ? v : 0.f;
            }
    __syncthreads();

    int rr = lane & 15, half = lane >> 4;
    u16* hp = (u16*)(ws + WS_HID);
    #pragma unroll
    for (int i = 0; i < 2; ++i)
        #pragma unroll
        for (int j = 0; j < 2; ++j) {
            u16x8 ov;
            #pragma unroll
            for (int q = 0; q < 8; ++q) ov[q] = f2bf(sm[w][i * 2 + j][rr * 16 + half * 8 + q]);
            *(u16x8*)(hp + (size_t)(row0 + i * 16 + rr) * DFF + (nb0 + j) * 16 + half * 8) = ov;
        }
}

// ---------------- K8: down GEMM, 32x32 per wave, +sum(bias), scatter to tokens ------
__global__ __launch_bounds__(128) void k_down_gemm(char* ws, const float* __restrict__ down_b,
                                                   float* __restrict__ out) {
    __shared__ float sm[4][4][256];
    int lane = threadIdx.x & 31, w = threadIdx.x >> 5;
    int m = blockIdx.y;
    int nb0 = (blockIdx.x * 4 + w) * 2;
    int row0 = m * 32;

    const bf16* A0 = (const bf16*)(ws + WS_HID) +
                     (size_t)(row0 + (lane & 15)) * DFF + (lane >> 4) * 16;
    const bf16* A1 = A0 + (size_t)16 * DFF;
    const bf16* Bp = (const bf16*)(ws + WS_WDS) + (size_t)nb0 * 512 + lane * 16;
    constexpr size_t BSTRIDE = (size_t)(DMODEL / 16) * 512;

    f32x8 acc[2][2];
    #pragma unroll
    for (int i = 0; i < 2; ++i)
        #pragma unroll
        for (int j = 0; j < 2; ++j) acc[i][j] = (f32x8){0.f,0.f,0.f,0.f,0.f,0.f,0.f,0.f};

    #pragma unroll 2
    for (int kb = 0; kb < DFF / 32; ++kb) {
        bf16x8 a00 = *(const bf16x8*)(A0 + kb * 32);
        bf16x8 a01 = *(const bf16x8*)(A0 + kb * 32 + 8);
        bf16x8 a10 = *(const bf16x8*)(A1 + kb * 32);
        bf16x8 a11 = *(const bf16x8*)(A1 + kb * 32 + 8);
        bf16x16 af0 = __builtin_shufflevector(a00, a01, 0,1,2,3,4,5,6,7,8,9,10,11,12,13,14,15);
        bf16x16 af1 = __builtin_shufflevector(a10, a11, 0,1,2,3,4,5,6,7,8,9,10,11,12,13,14,15);
        bf16x16 bf0 = *(const bf16x16*)(Bp + kb * BSTRIDE);
        bf16x16 bf1 = *(const bf16x16*)(Bp + kb * BSTRIDE + 512);
        acc[0][0] = __builtin_amdgcn_wmma_f32_16x16x32_bf16(false, af0, false, bf0, (short)0, acc[0][0], false, false);
        acc[0][1] = __builtin_amdgcn_wmma_f32_16x16x32_bf16(false, af0, false, bf1, (short)0, acc[0][1], false, false);
        acc[1][0] = __builtin_amdgcn_wmma_f32_16x16x32_bf16(false, af1, false, bf0, (short)0, acc[1][0], false, false);
        acc[1][1] = __builtin_amdgcn_wmma_f32_16x16x32_bf16(false, af1, false, bf1, (short)0, acc[1][1], false, false);
    }

    float bias[2];
    #pragma unroll
    for (int j = 0; j < 2; ++j) {
        float b = 0.f;
        #pragma unroll
        for (int t = 0; t < TILES; ++t) b += down_b[(size_t)t * DMODEL + (nb0 + j) * 16 + (lane & 15)];
        bias[j] = b;
    }

    #pragma unroll
    for (int i = 0; i < 2; ++i)
        #pragma unroll
        for (int j = 0; j < 2; ++j)
            #pragma unroll
            for (int r = 0; r < 8; ++r)
                sm[w][i * 2 + j][(r + 8 * (lane >> 4)) * 16 + (lane & 15)] = acc[i][j][r] + bias[j];
    __syncthreads();

    int rr = lane & 15, half = lane >> 4;
    const int* perm = (const int*)(ws + WS_PERM);
    #pragma unroll
    for (int i = 0; i < 2; ++i) {
        int tok = perm[row0 + i * 16 + rr];
        if (tok < 0) continue;
        #pragma unroll
        for (int j = 0; j < 2; ++j) {
            float* dst = out + (size_t)tok * DMODEL + (nb0 + j) * 16 + half * 8;
            float4 v0, v1;
            v0.x = sm[w][i*2+j][rr*16 + half*8 + 0]; v0.y = sm[w][i*2+j][rr*16 + half*8 + 1];
            v0.z = sm[w][i*2+j][rr*16 + half*8 + 2]; v0.w = sm[w][i*2+j][rr*16 + half*8 + 3];
            v1.x = sm[w][i*2+j][rr*16 + half*8 + 4]; v1.y = sm[w][i*2+j][rr*16 + half*8 + 5];
            v1.z = sm[w][i*2+j][rr*16 + half*8 + 6]; v1.w = sm[w][i*2+j][rr*16 + half*8 + 7];
            *(float4*)dst = v0;
            *(float4*)(dst + 4) = v1;
        }
    }
}

// ---------------- launch ----------------
extern "C" void kernel_launch(void* const* d_in, const int* in_sizes, int n_in,
                              void* d_out, int out_size, void* d_ws, size_t ws_size,
                              hipStream_t stream) {
    (void)in_sizes; (void)n_in; (void)out_size; (void)ws_size;
    const float* x      = (const float*)d_in[0];
    const float* gate_w = (const float*)d_in[1];
    const float* gate_b = (const float*)d_in[2];
    const float* up_w   = (const float*)d_in[3];
    const float* up_b   = (const float*)d_in[4];
    const float* down_w = (const float*)d_in[5];
    const float* down_b = (const float*)d_in[6];

    float* out      = (float*)d_out;
    float* gate_out = out + (size_t)NTOK * DMODEL;
    char* ws = (char*)d_ws;

    dim3 b256(256);
    k_init     <<<dim3((NPAD + 255) / 256), b256, 0, stream>>>(ws);
    k_gate     <<<dim3(NTOK), b256, 0, stream>>>(x, gate_w, gate_b, ws, gate_out);
    k_offsets  <<<dim3(1), dim3(1), 0, stream>>>(ws);
    k_scatter  <<<dim3(NTOK / 256), b256, 0, stream>>>(ws);
    k_xp       <<<dim3(NPAD), b256, 0, stream>>>(x, ws);
    k_pack_up  <<<dim3((TILES * (DMODEL / 2) * DFF) / 256), b256, 0, stream>>>(up_w, ws);
    k_pack_wds <<<dim3(((DFF / 2) * DMODEL) / 256), b256, 0, stream>>>(down_w, ws);
    k_up_gemm  <<<dim3(DFF / 128, MT), dim3(128), 0, stream>>>(ws, up_b);
    k_down_gemm<<<dim3(DMODEL / 128, MT), dim3(128), 0, stream>>>(ws, down_b, out);
}
